// LatentActionModel_43224550867377
// MI455X (gfx1250) — matrix operand order repaired and checked by reading, not compiled, not run on hardware
//
#include <hip/hip_runtime.h>
#include <hip/hip_bf16.h>
#include <math.h>

typedef __bf16 bf16_t;
typedef __attribute__((ext_vector_type(8)))  __bf16 v8bf;
typedef __attribute__((ext_vector_type(16))) __bf16 v16bf;
typedef __attribute__((ext_vector_type(8)))  float  v8f;
typedef int v4i_t __attribute__((vector_size(16)));            // matches builtin param
typedef __attribute__((address_space(3))) v4i_t* lds_v4i_p;    // LDS-qualified v4i*

#define B_ROWS   16384
#define FRAME    1024
#define HID      2048
#define ADIM     128
#define KCODES   1024
#define NLAYERS  3

// ---- WMMA GEMM tiling ----
#define BM 128
#define BN 128
#define BK 32
#define LDK 40   // padded LDS row stride in bf16 elems (80B rows -> conflict-free-ish)

// CDNA5 async global->LDS copy (ASYNCcnt-tracked).
#if __has_builtin(__builtin_amdgcn_global_load_async_to_lds_b128)
#define USE_ASYNC_LDS 1
#else
#define USE_ASYNC_LDS 0
#endif

#if USE_ASYNC_LDS
#define ASYNC_CP128(gsrc, ldst) \
  __builtin_amdgcn_global_load_async_to_lds_b128((v4i_t*)(gsrc), (lds_v4i_p)(ldst), 0, 0)
#if __has_builtin(__builtin_amdgcn_s_wait_asynccnt)
#define WAIT_ASYNC(n) __builtin_amdgcn_s_wait_asynccnt(n)
#else
#define WAIT_ASYNC(n) asm volatile("s_wait_asynccnt %0" :: "n"(n) : "memory")
#endif
#endif

// A fragment (16x32 bf16). ISA layout: lane<16 -> M=lane, K={0..7,16..23};
// lane>=16 -> M=lane-16, K={8..15,24..31}. Two aligned 16B LDS loads.
__device__ __forceinline__ v16bf frag_a(const bf16_t* base, int lane) {
  const int m  = lane & 15;
  const int ko = (lane >> 4) << 3;            // 0 or 8
  const bf16_t* p = base + m * LDK + ko;
  v8bf lo = *(const v8bf*)(p);
  v8bf hi = *(const v8bf*)(p + 16);
  return __builtin_shufflevector(lo, hi, 0,1,2,3,4,5,6,7,8,9,10,11,12,13,14,15);
}

// B fragment (32x16 bf16) from K-major Wt tile. ISA layout: lane<16 -> N=lane,
// K={0..7,16..23}; lane>=16 -> N=lane-16, K={8..15,24..31}.
__device__ __forceinline__ v16bf frag_b(const bf16_t* base, int lane) {
  const int n  = lane & 15;
  const int ko = (lane >> 4) << 3;            // 0 or 8
  const bf16_t* p = base + n * LDK + ko;
  v8bf lo = *(const v8bf*)(p);
  v8bf hi = *(const v8bf*)(p + 16);
  return __builtin_shufflevector(lo, hi, 0,1,2,3,4,5,6,7,8,9,10,11,12,13,14,15);
}

// C = A[M,K] * W[K,N] (+bias, +optional residual), A bf16 row-major,
// W pre-transposed to Wt[N,K] bf16. MODE: 0 f32-out, 1 bf16-out, 2 f32-out+residual.
template<int MODE>
__global__ __launch_bounds__(256)
void gemm_bf16_wmma(const bf16_t* __restrict__ A, const bf16_t* __restrict__ Bt,
                    const float* __restrict__ bias, const float* __restrict__ resid,
                    float* __restrict__ Cf, bf16_t* __restrict__ Cb,
                    int K, int N)
{
  const int tid  = threadIdx.x;
  const int lane = tid & 31;
  const int wid  = tid >> 5;      // 8 wave32 waves
  const int wm   = wid >> 1;      // 0..3 -> 32-row slab
  const int wn   = wid & 1;       // 0..1 -> 64-col slab
  const int m0   = blockIdx.y * BM;
  const int n0   = blockIdx.x * BN;

  // Per-thread staging coordinates: thread handles 4 (row, 8-elem-chunk) pairs.
  int srow[4], scol[4];
  #pragma unroll
  for (int it = 0; it < 4; ++it) {
    int id = it * 256 + tid;          // 0..1023
    srow[it] = id >> 2;               // 0..255
    scol[it] = (id & 3) << 3;         // 0,8,16,24
  }

  v8f acc[2][4];
  #pragma unroll
  for (int i = 0; i < 2; ++i)
    #pragma unroll
    for (int j = 0; j < 4; ++j)
      acc[i][j] = (v8f){0.f,0.f,0.f,0.f,0.f,0.f,0.f,0.f};

  const int nk = K / BK;

#if USE_ASYNC_LDS
  // -------- double-buffered async GLOBAL_LOAD_ASYNC_TO_LDS_B128 pipeline -----
  __shared__ __align__(16) bf16_t lA[2][BM * LDK];
  __shared__ __align__(16) bf16_t lB[2][BN * LDK];

  auto stage = [&](int buf, int k0) {
    #pragma unroll
    for (int it = 0; it < 4; ++it) {
      const int r = srow[it], c8 = scol[it];
      ASYNC_CP128(A  + (size_t)(m0 + r) * K + k0 + c8, &lA[buf][r * LDK + c8]);
      ASYNC_CP128(Bt + (size_t)(n0 + r) * K + k0 + c8, &lB[buf][r * LDK + c8]);
    }
  };

  stage(0, 0);                          // prologue: 8 async ops/thread in flight
  for (int kt = 0; kt < nk; ++kt) {
    const int cur = kt & 1;
    if (kt + 1 < nk) {                  // prefetch next tile, then wait only for cur
      stage(cur ^ 1, (kt + 1) * BK);
      WAIT_ASYNC(8);
    } else {
      WAIT_ASYNC(0);
    }
    __syncthreads();                    // all waves' async writes to cur visible

    v16bf af[2], bfg[4];
    #pragma unroll
    for (int i = 0; i < 2; ++i)
      af[i] = frag_a(&lA[cur][(wm * 32 + i * 16) * LDK], lane);
    #pragma unroll
    for (int j = 0; j < 4; ++j)
      bfg[j] = frag_b(&lB[cur][(wn * 64 + j * 16) * LDK], lane);

    #pragma unroll
    for (int i = 0; i < 2; ++i)
      #pragma unroll
      for (int j = 0; j < 4; ++j)
        acc[i][j] = __builtin_amdgcn_wmma_f32_16x16x32_bf16(
            false, af[i], false, bfg[j], (short)0, acc[i][j], false, false);

    __syncthreads();                    // cur fully consumed before it is restaged
  }
#else
  // -------- synchronous fallback: bulk register staging, one wait per tile ----
  __shared__ __align__(16) bf16_t lA1[BM * LDK];
  __shared__ __align__(16) bf16_t lB1[BN * LDK];

  for (int kt = 0; kt < nk; ++kt) {
    const int k0 = kt * BK;
    v8bf ra[4], rb[4];
    #pragma unroll
    for (int it = 0; it < 4; ++it) {    // 8 independent b128 loads in flight
      ra[it] = *(const v8bf*)&A [(size_t)(m0 + srow[it]) * K + k0 + scol[it]];
      rb[it] = *(const v8bf*)&Bt[(size_t)(n0 + srow[it]) * K + k0 + scol[it]];
    }
    if (k0 + BK < K) {                  // CDNA5 global_prefetch_b8 of next K-tile
      int r = tid >> 1;
      __builtin_prefetch(&A [(size_t)(m0 + r) * K + k0 + BK + (tid & 1) * 16], 0, 1);
      __builtin_prefetch(&Bt[(size_t)(n0 + r) * K + k0 + BK + (tid & 1) * 16], 0, 1);
    }
    __syncthreads();                    // previous iteration done reading LDS
    #pragma unroll
    for (int it = 0; it < 4; ++it) {
      *(v8bf*)&lA1[srow[it] * LDK + scol[it]] = ra[it];
      *(v8bf*)&lB1[srow[it] * LDK + scol[it]] = rb[it];
    }
    __syncthreads();

    v16bf af[2], bfg[4];
    #pragma unroll
    for (int i = 0; i < 2; ++i)
      af[i] = frag_a(&lA1[(wm * 32 + i * 16) * LDK], lane);
    #pragma unroll
    for (int j = 0; j < 4; ++j)
      bfg[j] = frag_b(&lB1[(wn * 64 + j * 16) * LDK], lane);

    #pragma unroll
    for (int i = 0; i < 2; ++i)
      #pragma unroll
      for (int j = 0; j < 4; ++j)
        acc[i][j] = __builtin_amdgcn_wmma_f32_16x16x32_bf16(
            false, af[i], false, bfg[j], (short)0, acc[i][j], false, false);
  }
#endif

  // Epilogue: C layout -> lane<16: N=lane, M=r; lane>=16: N=lane-16, M=8+r.
  const int mBase = m0 + wm * 32 + ((lane >> 4) << 3);
  const int nBase = n0 + wn * 64 + (lane & 15);
  #pragma unroll
  for (int i = 0; i < 2; ++i) {
    #pragma unroll
    for (int j = 0; j < 4; ++j) {
      const int nn = nBase + j * 16;
      const float bv = bias[nn];
      #pragma unroll
      for (int r = 0; r < 8; ++r) {
        const int mm = mBase + i * 16 + r;
        float v = acc[i][j][r] + bv;
        if (MODE == 2) v += resid[(size_t)mm * N + nn];
        if (MODE == 1) Cb[(size_t)mm * N + nn] = (bf16_t)v;
        else           Cf[(size_t)mm * N + nn] = v;
      }
    }
  }
}

// ---- LayerNorm + exact GELU, fp32 in -> bf16 out, one block per row ----
__global__ __launch_bounds__(256)
void ln_gelu_kernel(const float* __restrict__ X, const float* __restrict__ gamma,
                    const float* __restrict__ beta, bf16_t* __restrict__ Y, int N)
{
  const int row = blockIdx.x;
  const int tid = threadIdx.x;
  const float* x = X + (size_t)row * N;
  __shared__ float sm[256];

  float s = 0.f;
  for (int j = tid; j < N; j += 256) s += x[j];
  sm[tid] = s; __syncthreads();
  for (int st = 128; st > 0; st >>= 1) { if (tid < st) sm[tid] += sm[tid + st]; __syncthreads(); }
  const float mean = sm[0] / (float)N;
  __syncthreads();

  float v = 0.f;
  for (int j = tid; j < N; j += 256) { float d = x[j] - mean; v = fmaf(d, d, v); }
  sm[tid] = v; __syncthreads();
  for (int st = 128; st > 0; st >>= 1) { if (tid < st) sm[tid] += sm[tid + st]; __syncthreads(); }
  const float rstd = rsqrtf(sm[0] / (float)N + 1e-5f);

  bf16_t* y = Y + (size_t)row * N;
  for (int j = tid; j < N; j += 256) {
    float h = (x[j] - mean) * rstd * gamma[j] + beta[j];
    float g = 0.5f * h * (1.0f + erff(h * 0.70710678118654752f));
    y[j] = (bf16_t)g;
  }
}

// ---- VQ: argmin over codebook, gather, straight-through, stats ----
__global__ __launch_bounds__(256)
void vq_kernel(const float* __restrict__ z, const float* __restrict__ cb,
               float* __restrict__ idx_out, bf16_t* __restrict__ predIn,
               float* __restrict__ counts, float* __restrict__ vq_accum)
{
  const int b   = blockIdx.x;
  const int tid = threadIdx.x;
  __shared__ __align__(16) float4 zs[ADIM / 4];
  __shared__ float rdist[256];
  __shared__ int   ridx[256];

  if (tid < ADIM / 4) zs[tid] = ((const float4*)(z + (size_t)b * ADIM))[tid];
  __syncthreads();

  float best = 3.402823466e38f;
  int bi = 0;
  for (int k = tid; k < KCODES; k += 256) {
    const float4* c = (const float4*)(cb + (size_t)k * ADIM);
    float d = 0.f;
    #pragma unroll
    for (int j = 0; j < ADIM / 4; ++j) {
      float4 zv = zs[j], cv = c[j];
      float t0 = zv.x - cv.x, t1 = zv.y - cv.y, t2 = zv.z - cv.z, t3 = zv.w - cv.w;
      d = fmaf(t0, t0, d); d = fmaf(t1, t1, d); d = fmaf(t2, t2, d); d = fmaf(t3, t3, d);
    }
    if (d < best) { best = d; bi = k; }
  }
  rdist[tid] = best; ridx[tid] = bi; __syncthreads();
  for (int st = 128; st > 0; st >>= 1) {
    if (tid < st) {
      float od = rdist[tid + st]; int oi = ridx[tid + st];
      if (od < rdist[tid] || (od == rdist[tid] && oi < ridx[tid])) { rdist[tid] = od; ridx[tid] = oi; }
    }
    __syncthreads();
  }
  const int kbest = ridx[0];
  if (tid == 0) { idx_out[b] = (float)kbest; atomicAdd(&counts[kbest], 1.0f); }

  float loc = 0.f;
  if (tid < ADIM) {
    const float zz  = z[(size_t)b * ADIM + tid];
    const float q   = cb[(size_t)kbest * ADIM + tid];
    const float qst = zz + (q - zz);                       // straight-through value
    predIn[(size_t)b * (FRAME + ADIM) + FRAME + tid] = (bf16_t)qst;
    const float df = zz - q;
    loc = df * df;
  }
  rdist[tid] = loc; __syncthreads();
  for (int st = 128; st > 0; st >>= 1) { if (tid < st) rdist[tid] += rdist[tid + st]; __syncthreads(); }
  if (tid == 0) atomicAdd(vq_accum, rdist[0]);
}

// ---- fp32 W[K,N] -> bf16 Wt[N,K] tiled transpose-convert ----
__global__ __launch_bounds__(256)
void transpose_convert_kernel(const float* __restrict__ W, bf16_t* __restrict__ Wt,
                              int K, int N)
{
  __shared__ float tile[32][33];
  const int nb = blockIdx.x * 32;
  const int kb = blockIdx.y * 32;
  const int tx = threadIdx.x & 31;
  const int ty = threadIdx.x >> 5;     // 0..7
  #pragma unroll
  for (int i = 0; i < 32; i += 8)
    tile[ty + i][tx] = W[(size_t)(kb + ty + i) * N + nb + tx];
  __syncthreads();
  #pragma unroll
  for (int i = 0; i < 32; i += 8)
    Wt[(size_t)(nb + ty + i) * K + kb + tx] = (bf16_t)tile[tx][ty + i];
}

// ---- input concat + fp32->bf16 ----
__global__ void concat_convert_kernel(const float* __restrict__ ft, const float* __restrict__ ft1,
                                      bf16_t* __restrict__ act)
{
  const size_t i = (size_t)blockIdx.x * 256 + threadIdx.x;   // < B_ROWS*FRAME
  const size_t b = i >> 10;
  const size_t j = i & 1023;
  act[b * (2 * FRAME) + j]         = (bf16_t)ft[i];
  act[b * (2 * FRAME) + FRAME + j] = (bf16_t)ft1[i];
}

__global__ void predfill_kernel(const float* __restrict__ ft, bf16_t* __restrict__ predIn)
{
  const size_t i = (size_t)blockIdx.x * 256 + threadIdx.x;
  const size_t b = i >> 10;
  const size_t j = i & 1023;
  predIn[b * (FRAME + ADIM) + j] = (bf16_t)ft[i];
}

__global__ void zero_kernel(float* p, int n) {
  int i = blockIdx.x * 256 + threadIdx.x;
  if (i < n) p[i] = 0.f;
}

__global__ __launch_bounds__(256)
void recon_kernel(const float* __restrict__ pred, const float* __restrict__ tgt,
                  float* __restrict__ accum, int n)
{
  __shared__ float sm[256];
  float s = 0.f;
  for (size_t i = (size_t)blockIdx.x * 256 + threadIdx.x; i < (size_t)n;
       i += (size_t)gridDim.x * 256) {
    float d = pred[i] - tgt[i]; s = fmaf(d, d, s);
  }
  sm[threadIdx.x] = s; __syncthreads();
  for (int st = 128; st > 0; st >>= 1) { if (threadIdx.x < st) sm[threadIdx.x] += sm[threadIdx.x + st]; __syncthreads(); }
  if (threadIdx.x == 0) atomicAdd(accum, sm[0]);
}

__global__ __launch_bounds__(256)
void finalize_kernel(const float* __restrict__ counts, const float* __restrict__ accums,
                     float* __restrict__ out)
{
  __shared__ float sm[256];
  float s = 0.f;
  for (int k = threadIdx.x; k < KCODES; k += 256) {
    float p = counts[k] * (1.0f / (float)B_ROWS);
    s += p * logf(p + 1e-10f);
  }
  sm[threadIdx.x] = s; __syncthreads();
  for (int st = 128; st > 0; st >>= 1) { if (threadIdx.x < st) sm[threadIdx.x] += sm[threadIdx.x + st]; __syncthreads(); }
  if (threadIdx.x == 0) {
    const float mse_vq     = accums[0] / ((float)B_ROWS * (float)ADIM);
    const float commitment = mse_vq;
    const float codebook_l = mse_vq;
    const float vq_loss    = codebook_l + 0.25f * commitment;
    const float recon      = accums[1] / ((float)B_ROWS * (float)FRAME);
    out[0] = recon + vq_loss;   // total_loss
    out[1] = recon;             // reconstruction_loss
    out[2] = vq_loss;
    out[3] = commitment;
    out[4] = codebook_l;
    out[5] = expf(-sm[0]);      // perplexity
  }
}

extern "C" void kernel_launch(void* const* d_in, const int* in_sizes, int n_in,
                              void* d_out, int out_size, void* d_ws, size_t ws_size,
                              hipStream_t stream)
{
  const float* frame_t    = (const float*)d_in[0];
  const float* frame_t1   = (const float*)d_in[1];
  const float* enc_in_W   = (const float*)d_in[2];
  const float* enc_in_b   = (const float*)d_in[3];
  const float* enc_W      = (const float*)d_in[4];
  const float* enc_b      = (const float*)d_in[5];
  const float* enc_g      = (const float*)d_in[6];
  const float* enc_beta   = (const float*)d_in[7];
  const float* enc_out_W  = (const float*)d_in[8];
  const float* enc_out_b  = (const float*)d_in[9];
  const float* codebook   = (const float*)d_in[10];
  const float* pred_in_W  = (const float*)d_in[11];
  const float* pred_in_b  = (const float*)d_in[12];
  const float* pred_W     = (const float*)d_in[13];
  const float* pred_b     = (const float*)d_in[14];
  const float* pred_g     = (const float*)d_in[15];
  const float* pred_beta  = (const float*)d_in[16];
  const float* pred_out_W = (const float*)d_in[17];
  const float* pred_out_b = (const float*)d_in[18];

  float* out_pred   = (float*)d_out;                      // [B, FRAME]
  float* out_idx    = out_pred + (size_t)B_ROWS * FRAME;  // [B] (as float)
  float* out_scalar = out_idx + B_ROWS;                   // 6 scalars

  char* ws = (char*)d_ws;
  size_t off = 0;
  auto alloc = [&](size_t bytes) -> char* {
    char* p = ws + off;
    off += (bytes + 255) & ~(size_t)255;
    return p;
  };
  bf16_t* actIn    = (bf16_t*)alloc((size_t)B_ROWS * (2 * FRAME) * 2); // concat in; reused as predIn [B,1152]
  bf16_t* actB     = (bf16_t*)alloc((size_t)B_ROWS * HID * 2);         // bf16 activations
  float*  actF     = (float*) alloc((size_t)B_ROWS * HID * 4);         // pre-LN fp32
  float*  zbuf     = (float*) alloc((size_t)B_ROWS * ADIM * 4);        // z
  bf16_t* wEncIn   = (bf16_t*)alloc((size_t)(2 * FRAME) * HID * 2);
  bf16_t* wEnc     = (bf16_t*)alloc((size_t)NLAYERS * HID * HID * 2);
  bf16_t* wEncOut  = (bf16_t*)alloc((size_t)HID * ADIM * 2);
  bf16_t* wPredIn  = (bf16_t*)alloc((size_t)(FRAME + ADIM) * HID * 2);
  bf16_t* wPred    = (bf16_t*)alloc((size_t)NLAYERS * HID * HID * 2);
  bf16_t* wPredOut = (bf16_t*)alloc((size_t)HID * FRAME * 2);
  float*  counts   = (float*) alloc(KCODES * 4);
  float*  accums   = (float*) alloc(2 * 4);   // [0]=vq sum, [1]=recon sum

  const dim3 blk(256);
  const int elemBlocks = (B_ROWS * FRAME) / 256;   // 65536

  // 0) zero stat accumulators
  zero_kernel<<<dim3((KCODES + 255) / 256), blk, 0, stream>>>(counts, KCODES);
  zero_kernel<<<dim3(1), blk, 0, stream>>>(accums, 2);

  // 1) inputs -> bf16, concat [frame_t | frame_t1]
  concat_convert_kernel<<<dim3(elemBlocks), blk, 0, stream>>>(frame_t, frame_t1, actIn);

  // 2) weights: fp32 [K,N] -> bf16 [N,K]
  transpose_convert_kernel<<<dim3(HID / 32, (2 * FRAME) / 32), blk, 0, stream>>>(enc_in_W, wEncIn, 2 * FRAME, HID);
  for (int l = 0; l < NLAYERS; ++l)
    transpose_convert_kernel<<<dim3(HID / 32, HID / 32), blk, 0, stream>>>(
        enc_W + (size_t)l * HID * HID, wEnc + (size_t)l * HID * HID, HID, HID);
  transpose_convert_kernel<<<dim3(ADIM / 32, HID / 32), blk, 0, stream>>>(enc_out_W, wEncOut, HID, ADIM);
  transpose_convert_kernel<<<dim3(HID / 32, (FRAME + ADIM) / 32), blk, 0, stream>>>(pred_in_W, wPredIn, FRAME + ADIM, HID);
  for (int l = 0; l < NLAYERS; ++l)
    transpose_convert_kernel<<<dim3(HID / 32, HID / 32), blk, 0, stream>>>(
        pred_W + (size_t)l * HID * HID, wPred + (size_t)l * HID * HID, HID, HID);
  transpose_convert_kernel<<<dim3(FRAME / 32, HID / 32), blk, 0, stream>>>(pred_out_W, wPredOut, HID, FRAME);

  // 3) encoder input GEMM: [B,2048]x[2048,2048] + b -> bf16 actB
  gemm_bf16_wmma<1><<<dim3(HID / BN, B_ROWS / BM), blk, 0, stream>>>(
      actIn, wEncIn, enc_in_b, nullptr, nullptr, actB, 2 * FRAME, HID);

  // 4) 3x [GEMM -> LN -> GELU]
  for (int l = 0; l < NLAYERS; ++l) {
    gemm_bf16_wmma<0><<<dim3(HID / BN, B_ROWS / BM), blk, 0, stream>>>(
        actB, wEnc + (size_t)l * HID * HID, enc_b + (size_t)l * HID, nullptr, actF, nullptr, HID, HID);
    ln_gelu_kernel<<<dim3(B_ROWS), blk, 0, stream>>>(
        actF, enc_g + (size_t)l * HID, enc_beta + (size_t)l * HID, actB, HID);
  }

  // 5) z = actB @ enc_out_W + b  -> fp32
  gemm_bf16_wmma<0><<<dim3(ADIM / BN, B_ROWS / BM), blk, 0, stream>>>(
      actB, wEncOut, enc_out_b, nullptr, zbuf, nullptr, HID, ADIM);

  // 6) predictor input = [frame_t | q_st] in bf16 (actIn reused, safe: enc_in GEMM done)
  predfill_kernel<<<dim3(elemBlocks), blk, 0, stream>>>(frame_t, actIn);
  vq_kernel<<<dim3(B_ROWS), blk, 0, stream>>>(zbuf, codebook, out_idx, actIn, counts, accums);

  // 7) predictor input GEMM: [B,1152]x[1152,2048] -> bf16 actB
  gemm_bf16_wmma<1><<<dim3(HID / BN, B_ROWS / BM), blk, 0, stream>>>(
      actIn, wPredIn, pred_in_b, nullptr, nullptr, actB, FRAME + ADIM, HID);

  // 8) 3x [GEMM -> LN -> GELU]
  for (int l = 0; l < NLAYERS; ++l) {
    gemm_bf16_wmma<0><<<dim3(HID / BN, B_ROWS / BM), blk, 0, stream>>>(
        actB, wPred + (size_t)l * HID * HID, pred_b + (size_t)l * HID, nullptr, actF, nullptr, HID, HID);
    ln_gelu_kernel<<<dim3(B_ROWS), blk, 0, stream>>>(
        actF, pred_g + (size_t)l * HID, pred_beta + (size_t)l * HID, actB, HID);
  }

  // 9) pred_frame = frame_t + actB @ pred_out_W + b  (fp32 straight into d_out)
  gemm_bf16_wmma<2><<<dim3(FRAME / BN, B_ROWS / BM), blk, 0, stream>>>(
      actB, wPredOut, pred_out_b, frame_t, out_pred, nullptr, HID, FRAME);

  // 10) losses + perplexity
  recon_kernel<<<dim3(2048), blk, 0, stream>>>(out_pred, frame_t1, accums + 1, B_ROWS * FRAME);
  finalize_kernel<<<dim3(1), blk, 0, stream>>>(counts, accums, out_scalar);
}
